// HAGCM_58763742544854
// MI455X (gfx1250) — matrix operand order repaired and checked by reading, not compiled, not run on hardware
//
#include <hip/hip_runtime.h>
#include <hip/hip_bf16.h>

typedef _Float16 v16h __attribute__((ext_vector_type(16)));
typedef float    v8f  __attribute__((ext_vector_type(8)));

#define N_NODES 100000
#define N_EDGES 1600000
#define BB 32
#define SS 512
#define DD 768
#define HH 128
#define GG 64

// ---------------- attention scores: tanh(x@W1+b1)@w2+b2 ----------------
__global__ void attn_scores_kernel(const float* __restrict__ hx,
                                   const float* __restrict__ w1,
                                   const float* __restrict__ b1,
                                   const float* __restrict__ w2,
                                   const float* __restrict__ b2,
                                   float* __restrict__ scores) {
    __shared__ float xs[DD];
    __shared__ float red[HH];
    int row = blockIdx.x;                 // b*S + s
    int t = threadIdx.x;                  // 0..127
    const float* xr = hx + (size_t)row * DD;
    for (int d = t; d < DD; d += HH) xs[d] = xr[d];
    __syncthreads();
    float acc = 0.f;
    for (int d = 0; d < DD; ++d) acc = fmaf(xs[d], w1[d * HH + t], acc);
    float h = tanhf(acc + b1[t]);
    red[t] = h * w2[t];
    __syncthreads();
    for (int s = HH / 2; s > 0; s >>= 1) {
        if (t < s) red[t] += red[t + s];
        __syncthreads();
    }
    if (t == 0) scores[row] = red[0] + b2[0];
}

// ---------------- softmax over S + weighted pooling ----------------
__global__ void softmax_pool_kernel(const float* __restrict__ hx,
                                    const float* __restrict__ scores,
                                    float* __restrict__ hpool) {
    __shared__ float w[SS];
    __shared__ float red[256];
    int b = blockIdx.x, t = threadIdx.x;  // 256 threads
    for (int s = t; s < SS; s += 256) w[s] = scores[b * SS + s];
    __syncthreads();
    red[t] = fmaxf(w[t], w[t + 256]);
    __syncthreads();
    for (int s = 128; s > 0; s >>= 1) {
        if (t < s) red[t] = fmaxf(red[t], red[t + s]);
        __syncthreads();
    }
    float mx = red[0];
    __syncthreads();
    float e0 = expf(w[t] - mx), e1 = expf(w[t + 256] - mx);
    w[t] = e0; w[t + 256] = e1;
    red[t] = e0 + e1;
    __syncthreads();
    for (int s = 128; s > 0; s >>= 1) {
        if (t < s) red[t] += red[t + s];
        __syncthreads();
    }
    float inv = 1.f / red[0];
    for (int d = t; d < DD; d += 256) {
        float acc = 0.f;
        const float* xp = hx + (size_t)b * SS * DD + d;
        for (int s = 0; s < SS; ++s) acc = fmaf(w[s], xp[(size_t)s * DD], acc);
        hpool[b * DD + d] = acc * inv;
    }
}

// ---------------- txt = hpool @ reduce_text_w + b ----------------
__global__ void txt_kernel(const float* __restrict__ hpool,
                           const float* __restrict__ rtw,
                           const float* __restrict__ rtb,
                           float* __restrict__ txt) {
    int b = blockIdx.x, j = threadIdx.x;  // 64 threads
    float acc = rtb[j];
    const float* hp = hpool + b * DD;
    for (int d = 0; d < DD; ++d) acc = fmaf(hp[d], rtw[d * GG + j], acc);
    txt[b * GG + j] = acc;
}

// ------- normalize txt rows + pre-swizzle into WMMA A-fragments -------
// frag idx = ((mt*2 + ks)*32 + lane)*16 + j ; A layout: k = 16*(lane/16) + j (contiguous)
__global__ void txt_frag_kernel(const float* __restrict__ txt, _Float16* __restrict__ frag) {
    __shared__ float tl[BB * GG];
    __shared__ float invn[BB];
    int t = threadIdx.x;  // 256
    for (int i = t; i < BB * GG; i += 256) tl[i] = txt[i];
    __syncthreads();
    if (t < BB) {
        float ss = 0.f;
        for (int c = 0; c < GG; ++c) { float v = tl[t * GG + c]; ss = fmaf(v, v, ss); }
        invn[t] = 1.f / fmaxf(sqrtf(ss), 1e-12f);
    }
    __syncthreads();
    for (int idx = t; idx < 2048; idx += 256) {
        int j = idx & 15, lane = (idx >> 4) & 31, rest = idx >> 9;  // rest = mt*2+ks
        int ks = rest & 1, mt = rest >> 1;
        int m = mt * 16 + (lane & 15);
        int k = ks * 32 + j + ((lane >> 4) << 4);
        frag[idx] = (_Float16)(tl[m * GG + k] * invn[m]);
    }
}

// ------- pre-swizzle a [K,N] f32 weight into WMMA B-fragments (f16) -------
// B layout: lanes 0-15 hold K=0..7,16..23, lanes 16-31 hold K=8..15,24..31; n = lane&15
__global__ void bswz_kernel(const float* __restrict__ W, _Float16* __restrict__ frag,
                            int Ndim, int total) {
    int idx = blockIdx.x * 256 + threadIdx.x;
    if (idx >= total) return;
    int j = idx & 15, lane = (idx >> 4) & 31, rest = idx >> 9;
    int NT = Ndim >> 4;
    int nt = rest % NT, ks = rest / NT;
    int k = ks * 32 + (j & 7) + ((j >> 3) << 4) + ((lane >> 4) << 3);
    int n = nt * 16 + (lane & 15);
    frag[idx] = (_Float16)W[k * Ndim + n];
}

// ------- WMMA GEMM: C[M, NT*16] = act(A[M, KSTEPS*32]) @ Bswz (+bias) -------
template <int NT, int KSTEPS, bool RELU>
__global__ void gemm_wmma_kernel(const float* __restrict__ A,
                                 const _Float16* __restrict__ Bsw,
                                 const float* __restrict__ bias,
                                 float* __restrict__ C, int Mtiles) {
    int lane = threadIdx.x & 31;
    int mt = blockIdx.x * 8 + (threadIdx.x >> 5);
    if (mt >= Mtiles) return;                     // wave-uniform: EXEC stays all-1
    const int Kdim = KSTEPS * 32, Ndim = NT * 16;
    int row = mt * 16 + (lane & 15);
    const float* arow = A + (size_t)row * Kdim;
    int khi = (lane >> 4) << 3;                   // 0 or 8
    v8f zero = {};
    v8f acc[NT];
#pragma unroll
    for (int nt = 0; nt < NT; ++nt) acc[nt] = zero;
#pragma unroll
    for (int ks = 0; ks < KSTEPS; ++ks) {
        v16h a;
        const float* ap = arow + ks * 32 + khi;
#pragma unroll
        for (int u = 0; u < 8; ++u) {
            float f0 = ap[u], f1 = ap[16 + u];
            if (RELU) { f0 = fmaxf(f0, 0.f); f1 = fmaxf(f1, 0.f); }
            a[u] = (_Float16)f0;
            a[8 + u] = (_Float16)f1;
        }
#pragma unroll
        for (int nt = 0; nt < NT; ++nt) {
            v16h bf = *(const v16h*)(Bsw + (((size_t)ks * NT + nt) * 32 + lane) * 16);
            acc[nt] = __builtin_amdgcn_wmma_f32_16x16x32_f16(
                false, a, false, bf, (short)0, acc[nt], false, false);
        }
    }
    int rowc = mt * 16 + khi;
#pragma unroll
    for (int nt = 0; nt < NT; ++nt) {
        int col = nt * 16 + (lane & 15);
        float bv = bias ? bias[col] : 0.f;
#pragma unroll
        for (int r = 0; r < 8; ++r)
            C[(size_t)(rowc + r) * Ndim + col] = acc[nt][r] + bv;
    }
}

// ---------------- degree / normalization ----------------
__global__ void deg_init_kernel(float* deg, int n) {
    int i = blockIdx.x * 256 + threadIdx.x;
    if (i < n) deg[i] = 1.f;  // self loop
}
__global__ void deg_edges_kernel(const int* __restrict__ dst, float* deg, int e) {
    int i = blockIdx.x * 256 + threadIdx.x;
    if (i < e) atomicAdd(deg + dst[i], 1.f);
}
__global__ void dinv_kernel(const float* __restrict__ deg, float* dinv, int n) {
    int i = blockIdx.x * 256 + threadIdx.x;
    if (i < n) dinv[i] = rsqrtf(deg[i]);
}

// ------- conv init: out = bias + y * dinv^2 (self-loop term) -------
__global__ void conv_init_kernel(const float* __restrict__ y, const float* __restrict__ dinv,
                                 const float* __restrict__ bias, float* __restrict__ out,
                                 int total) {
    int i = blockIdx.x * 256 + threadIdx.x;
    if (i >= total) return;
    int node = i >> 6, c = i & 63;
    float di = dinv[node];
    out[i] = fmaf(y[i], di * di, bias[c]);
}

// ------- edge scatter: out[dst] += y[src] * dinv[src]*dinv[dst] -------
__global__ void edge_scatter_kernel(const int* __restrict__ src, const int* __restrict__ dst,
                                    const float* __restrict__ y, const float* __restrict__ dinv,
                                    float* __restrict__ out, int nEdges) {
    int t = blockIdx.x * 256 + threadIdx.x;
    int e = t >> 6;
    if (e >= nEdges) return;
    int c = t & 63;
    int s = src[e], d = dst[e];
    float w = dinv[s] * dinv[d];
    atomicAdd(out + (size_t)d * GG + c, y[(size_t)s * GG + c] * w);
}

// ------- row-normalize conv2 output into f16 (B operand of final GEMM) -------
__global__ void norm_rows_f16_kernel(const float* __restrict__ x, _Float16* __restrict__ xn,
                                     int nRows) {
    int lane = threadIdx.x & 31;
    int r = blockIdx.x * 8 + (threadIdx.x >> 5);
    if (r >= nRows) return;
    const float* xp = x + (size_t)r * GG;
    float a = xp[lane], b = xp[lane + 32];
    float ss = a * a + b * b;
    for (int off = 16; off > 0; off >>= 1) ss += __shfl_xor(ss, off, 32);
    float s = 1.f / fmaxf(sqrtf(ss), 1e-12f);
    _Float16* xo = xn + (size_t)r * GG;
    xo[lane] = (_Float16)(a * s);
    xo[lane + 32] = (_Float16)(b * s);
}

// ------- final GEMM: out[32,100000] = txt_n @ xn^T via WMMA -------
__global__ void final_gemm_kernel(const _Float16* __restrict__ afrag,
                                  const _Float16* __restrict__ xn,
                                  float* __restrict__ out, int nTiles) {
    int lane = threadIdx.x & 31;
    int nt = blockIdx.x * 8 + (threadIdx.x >> 5);
    if (nt >= nTiles) return;
    v8f acc0 = {}, acc1 = {};
    int node = nt * 16 + (lane & 15);
    const _Float16* xrow = xn + (size_t)node * GG;
    int kb = (lane >> 4) << 4;  // B layout: upper 16 lanes hold K+16
#pragma unroll
    for (int ks = 0; ks < 2; ++ks) {
        v16h bf = *(const v16h*)(xrow + ks * 32 + kb);
        v16h a0 = *(const v16h*)(afrag + ((size_t)(0 * 2 + ks) * 32 + lane) * 16);
        v16h a1 = *(const v16h*)(afrag + ((size_t)(1 * 2 + ks) * 32 + lane) * 16);
        acc0 = __builtin_amdgcn_wmma_f32_16x16x32_f16(false, a0, false, bf, (short)0, acc0, false, false);
        acc1 = __builtin_amdgcn_wmma_f32_16x16x32_f16(false, a1, false, bf, (short)0, acc1, false, false);
    }
    int rb = (lane >> 4) << 3;
    int col = nt * 16 + (lane & 15);
#pragma unroll
    for (int r = 0; r < 8; ++r) {
        out[(size_t)(rb + r) * N_NODES + col] = acc0[r];
        out[(size_t)(16 + rb + r) * N_NODES + col] = acc1[r];
    }
}

// ---------------- workspace layout (bytes, all 128-aligned) ----------------
static const size_t OFF_LAB    = 0;          // 51,200,000  f32 [N,128]  (reused as y2 [N,64])
static const size_t OFF_Y1     = 51200000;   // 25,600,000  f32 [N,64]   (reused as out2)
static const size_t OFF_OUT1   = 76800000;   // 25,600,000  f32 [N,64]
static const size_t OFF_XN     = 102400000;  // 12,800,000  f16 [N,64]
static const size_t OFF_DEG    = 115200000;  //    400,000
static const size_t OFF_DINV   = 115600000;  //    400,000
static const size_t OFF_SCORES = 116000000;  //     65,536
static const size_t OFF_HPOOL  = 116065536;  //     98,304
static const size_t OFF_TXT    = 116163840;  //      8,192
static const size_t OFF_BSWR   = 116172032;  //    196,608  (24*8*512 f16)
static const size_t OFF_BSW1   = 116368640;  //     16,384  (4*4*512 f16)
static const size_t OFF_BSW2   = 116385024;  //      8,192  (2*4*512 f16)
static const size_t OFF_TXTF   = 116393216;  //      4,096  (4*512 f16)

extern "C" void kernel_launch(void* const* d_in, const int* in_sizes, int n_in,
                              void* d_out, int out_size, void* d_ws, size_t ws_size,
                              hipStream_t stream) {
    const float* hx   = (const float*)d_in[0];
    const float* lab0 = (const float*)d_in[1];
    const int*   ei   = (const int*)d_in[2];
    const float* aw1  = (const float*)d_in[3];
    const float* ab1  = (const float*)d_in[4];
    const float* aw2  = (const float*)d_in[5];
    const float* ab2  = (const float*)d_in[6];
    const float* rdw  = (const float*)d_in[7];
    const float* rdb  = (const float*)d_in[8];
    const float* rtw  = (const float*)d_in[9];
    const float* rtb  = (const float*)d_in[10];
    const float* g1w  = (const float*)d_in[11];
    const float* g1b  = (const float*)d_in[12];
    const float* g2w  = (const float*)d_in[13];
    const float* g2b  = (const float*)d_in[14];
    float* out = (float*)d_out;

    char* ws = (char*)d_ws;
    float*    lab    = (float*)(ws + OFF_LAB);
    float*    y1     = (float*)(ws + OFF_Y1);
    float*    out1   = (float*)(ws + OFF_OUT1);
    float*    y2     = (float*)(ws + OFF_LAB);   // alias lab (lab dead after y1)
    float*    out2   = (float*)(ws + OFF_Y1);    // alias y1 (y1 dead after out1)
    _Float16* xn     = (_Float16*)(ws + OFF_XN);
    float*    deg    = (float*)(ws + OFF_DEG);
    float*    dinv   = (float*)(ws + OFF_DINV);
    float*    scores = (float*)(ws + OFF_SCORES);
    float*    hpool  = (float*)(ws + OFF_HPOOL);
    float*    txt    = (float*)(ws + OFF_TXT);
    _Float16* bswr   = (_Float16*)(ws + OFF_BSWR);
    _Float16* bsw1   = (_Float16*)(ws + OFF_BSW1);
    _Float16* bsw2   = (_Float16*)(ws + OFF_BSW2);
    _Float16* txtf   = (_Float16*)(ws + OFF_TXTF);

    const int nedges = in_sizes[2] / 2;
    const int* esrc = ei;
    const int* edst = ei + nedges;
    const int mtiles = N_NODES / 16;              // 6250
    const int gemmBlocks = (mtiles + 7) / 8;      // 782

    // 1) attention pooling path
    attn_scores_kernel<<<BB * SS, HH, 0, stream>>>(hx, aw1, ab1, aw2, ab2, scores);
    softmax_pool_kernel<<<BB, 256, 0, stream>>>(hx, scores, hpool);
    txt_kernel<<<BB, GG, 0, stream>>>(hpool, rtw, rtb, txt);
    txt_frag_kernel<<<1, 256, 0, stream>>>(txt, txtf);

    // 2) pre-swizzle weights into WMMA B-fragments
    bswz_kernel<<<(24 * 8 * 512) / 256, 256, 0, stream>>>(rdw, bswr, HH, 24 * 8 * 512);
    bswz_kernel<<<(4 * 4 * 512) / 256, 256, 0, stream>>>(g1w, bsw1, GG, 4 * 4 * 512);
    bswz_kernel<<<(2 * 4 * 512) / 256, 256, 0, stream>>>(g2w, bsw2, GG, 2 * 4 * 512);

    // 3) lab = label_embeddings @ reduce_dim_w + b   [N,128]  (big WMMA GEMM)
    gemm_wmma_kernel<8, 24, false><<<gemmBlocks, 256, 0, stream>>>(lab0, bswr, rdb, lab, mtiles);

    // 4) degree + D^{-1/2}
    deg_init_kernel<<<(N_NODES + 255) / 256, 256, 0, stream>>>(deg, N_NODES);
    deg_edges_kernel<<<(nedges + 255) / 256, 256, 0, stream>>>(edst, deg, nedges);
    dinv_kernel<<<(N_NODES + 255) / 256, 256, 0, stream>>>(deg, dinv, N_NODES);

    // 5) GCN conv 1: y1 = lab @ gcn1_w ; out1 = bias + self + scatter
    gemm_wmma_kernel<4, 4, false><<<gemmBlocks, 256, 0, stream>>>(lab, bsw1, nullptr, y1, mtiles);
    conv_init_kernel<<<(N_NODES * GG) / 256, 256, 0, stream>>>(y1, dinv, g1b, out1, N_NODES * GG);
    edge_scatter_kernel<<<(nedges * GG) / 256, 256, 0, stream>>>(esrc, edst, y1, dinv, out1, nedges);

    // 6) GCN conv 2 (relu fused into A-read): y2 = relu(out1) @ gcn2_w
    gemm_wmma_kernel<4, 2, true><<<gemmBlocks, 256, 0, stream>>>(out1, bsw2, nullptr, y2, mtiles);
    conv_init_kernel<<<(N_NODES * GG) / 256, 256, 0, stream>>>(y2, dinv, g2b, out2, N_NODES * GG);
    edge_scatter_kernel<<<(nedges * GG) / 256, 256, 0, stream>>>(esrc, edst, y2, dinv, out2, nedges);

    // 7) normalize rows -> f16, final similarity GEMM via WMMA
    norm_rows_f16_kernel<<<(N_NODES + 7) / 8, 256, 0, stream>>>(out2, xn, N_NODES);
    final_gemm_kernel<<<gemmBlocks, 256, 0, stream>>>(txtf, xn, out, mtiles);

    (void)n_in; (void)out_size; (void)ws_size;
}
